// TransformerBlock_12936441495630
// MI455X (gfx1250) — matrix operand order repaired and checked by reading, not compiled, not run on hardware
//
#include <hip/hip_runtime.h>
#include <hip/hip_bf16.h>

typedef __bf16 bf16;
typedef __attribute__((ext_vector_type(16))) __bf16 bf16x16;
typedef __attribute__((ext_vector_type(8)))  __bf16 bf16x8;
typedef __attribute__((ext_vector_type(8)))  float  f32x8;

#define B_   4
#define S_   2048
#define DM   1024
#define NH   16
#define DK   64
#define DFF  2752
#define NEG_INF (-__builtin_inff())

// ---------------------------------------------------------------------------
// WMMA helper: D = A(16x32 bf16) * B(32x16 bf16) + C(16x16 f32)
// ---------------------------------------------------------------------------
static __device__ __forceinline__ f32x8 wmma_bf16(bf16x16 a, bf16x16 b, f32x8 c) {
  return __builtin_amdgcn_wmma_f32_16x16x32_bf16(false, a, false, b, (short)0, c,
                                                 false, false);
}

// Fast hardware reciprocal (v_rcp_f32)
static __device__ __forceinline__ float fast_rcp(float x) {
  return __builtin_amdgcn_rcpf(x);
}

// A-fragment (16x32, row-major source, lda in elements).
// Order used here (matches load_b_frag): lanes 0-15 row M=lane, K 0-15 contiguous; lanes 16-31 same rows, K 16-31.
static __device__ __forceinline__ bf16x16
load_a_frag(const bf16* A, int lda, int r0, int k0, int lane) {
  int r  = r0 + (lane & 15);
  int kb = k0 + ((lane >> 4) << 4);            // +0 or +16
  return *(const bf16x16*)(A + (size_t)r * lda + kb);
}

// B-fragment (32x16) from W stored [N,K] row-major (ldw in elements).
// Order used here (matches load_a_frag): lanes 0-15 col N=lane, K 0-15 contiguous; lanes 16-31 K 16-31.
static __device__ __forceinline__ bf16x16
load_b_frag(const bf16* W, int ldw, int n0, int k0, int lane) {
  int n  = n0 + (lane & 15);
  int kb = k0 + ((lane >> 4) << 4);            // +0 or +16
  return *(const bf16x16*)(W + (size_t)n * ldw + kb);
}

// ---------------------------------------------------------------------------
// f32 -> bf16 conversion
// ---------------------------------------------------------------------------
__global__ __launch_bounds__(256) void cvt_f32_bf16_kernel(
    const float* __restrict__ in, bf16* __restrict__ out, long n) {
  long i = (long)blockIdx.x * 256 + threadIdx.x;
  if (i < n) out[i] = (bf16)in[i];
}

// ---------------------------------------------------------------------------
// RMSNorm: one block (256 thr, 8 waves) per row of D=1024, bf16 output
// ---------------------------------------------------------------------------
__global__ __launch_bounds__(256) void rmsnorm_bf16_kernel(
    const float* __restrict__ x, const float* __restrict__ gain,
    bf16* __restrict__ out) {
  int row = blockIdx.x;
  const float* xr = x + (size_t)row * DM;
  float ss = 0.f;
#pragma unroll
  for (int i = threadIdx.x; i < DM; i += 256) { float v = xr[i]; ss += v * v; }
#pragma unroll
  for (int m = 1; m < 32; m <<= 1) ss += __shfl_xor(ss, m, 32);
  __shared__ float red[8];
  if ((threadIdx.x & 31) == 0) red[threadIdx.x >> 5] = ss;
  __syncthreads();
  float tot = 0.f;
#pragma unroll
  for (int w = 0; w < 8; ++w) tot += red[w];
  float rinv = rsqrtf(tot * (1.0f / DM) + 1e-5f);
  for (int i = threadIdx.x; i < DM; i += 256)
    out[(size_t)row * DM + i] = (bf16)(xr[i] * gain[i] * rinv);
}

// ---------------------------------------------------------------------------
// Generic WMMA GEMM: C[M,N] = A[M,K](bf16) * W[N,K]^T(bf16) (+ resid), f32 out
// block = 256 thr (8 waves), block tile 128x128, wave tile 32x64, K-step 32
// ---------------------------------------------------------------------------
template <bool RESID>
__global__ __launch_bounds__(256) void wmma_gemm_kernel(
    const bf16* __restrict__ A, const bf16* __restrict__ W,
    float* __restrict__ C, const float* __restrict__ resid,
    int M, int N, int K) {
  int lane = threadIdx.x & 31, wave = threadIdx.x >> 5;
  int m0 = blockIdx.y * 128 + (wave >> 1) * 32;
  int n0 = blockIdx.x * 128 + (wave & 1) * 64;
  f32x8 acc[2][4] = {};
  for (int k0 = 0; k0 < K; k0 += 32) {
    bf16x16 a[2];
#pragma unroll
    for (int i = 0; i < 2; ++i) a[i] = load_a_frag(A, K, m0 + i * 16, k0, lane);
    bf16x16 b[4];
#pragma unroll
    for (int t = 0; t < 4; ++t) b[t] = load_b_frag(W, K, n0 + t * 16, k0, lane);
#pragma unroll
    for (int i = 0; i < 2; ++i)
#pragma unroll
      for (int t = 0; t < 4; ++t) acc[i][t] = wmma_bf16(a[i], b[t], acc[i][t]);
  }
  int hf = lane >> 4, col = lane & 15;
#pragma unroll
  for (int i = 0; i < 2; ++i)
#pragma unroll
    for (int t = 0; t < 4; ++t)
#pragma unroll
      for (int j = 0; j < 8; ++j) {
        int m = m0 + i * 16 + j + hf * 8;
        int n = n0 + t * 16 + col;
        float v = acc[i][t][j];
        if (RESID) v += resid[(size_t)m * N + n];
        C[(size_t)m * N + n] = v;
      }
}

// ---------------------------------------------------------------------------
// Fused SwiGLU GEMM: HG[M,N] = bf16( silu(A*W1^T) * (A*W3^T) )
// block tile 128x64, wave tile 32x32, shared A fragments, two accumulators
// ---------------------------------------------------------------------------
__global__ __launch_bounds__(256) void wmma_gemm_swiglu_kernel(
    const bf16* __restrict__ A, const bf16* __restrict__ W1,
    const bf16* __restrict__ W3, bf16* __restrict__ HG,
    int M, int N, int K) {
  int lane = threadIdx.x & 31, wave = threadIdx.x >> 5;
  int m0 = blockIdx.y * 128 + (wave >> 1) * 32;
  int n0 = blockIdx.x * 64 + (wave & 1) * 32;
  f32x8 acch[2][2] = {};
  f32x8 accg[2][2] = {};
  for (int k0 = 0; k0 < K; k0 += 32) {
    bf16x16 a[2];
#pragma unroll
    for (int i = 0; i < 2; ++i) a[i] = load_a_frag(A, K, m0 + i * 16, k0, lane);
    bf16x16 b1[2], b3[2];
#pragma unroll
    for (int t = 0; t < 2; ++t) {
      b1[t] = load_b_frag(W1, K, n0 + t * 16, k0, lane);
      b3[t] = load_b_frag(W3, K, n0 + t * 16, k0, lane);
    }
#pragma unroll
    for (int i = 0; i < 2; ++i)
#pragma unroll
      for (int t = 0; t < 2; ++t) {
        acch[i][t] = wmma_bf16(a[i], b1[t], acch[i][t]);
        accg[i][t] = wmma_bf16(a[i], b3[t], accg[i][t]);
      }
  }
  int hf = lane >> 4, col = lane & 15;
#pragma unroll
  for (int i = 0; i < 2; ++i)
#pragma unroll
    for (int t = 0; t < 2; ++t)
#pragma unroll
      for (int j = 0; j < 8; ++j) {
        int m = m0 + i * 16 + j + hf * 8;
        int n = n0 + t * 16 + col;
        float h = acch[i][t][j];
        float g = accg[i][t][j];
        // silu(h) = h * sigmoid(h); v_rcp_f32 instead of IEEE divide chain
        float sw = h * fast_rcp(1.f + __expf(-h));
        HG[(size_t)m * N + n] = (bf16)(sw * g);
      }
}

// ---------------------------------------------------------------------------
// RoPE + QKV split. qkv f32 [B*S, 3*DM] with n = t*DM + h*DK + dk.
// Q scaled by 1/sqrt(DK); Q,K -> [bh, s, dk] bf16; V -> transposed [bh, dk, s].
// One thread per (b,h,s, dk-pair).
// ---------------------------------------------------------------------------
__global__ __launch_bounds__(256) void rope_split_kernel(
    const float* __restrict__ qkv, const int* __restrict__ pos,
    bf16* __restrict__ Qb, bf16* __restrict__ Kb, bf16* __restrict__ Vt) {
  long idx = (long)blockIdx.x * 256 + threadIdx.x;   // B*NH*S*32 total
  int  i   = (int)(idx & 31);                        // pair index 0..31
  long r   = idx >> 5;
  int  s   = (int)(r & (S_ - 1));
  int  bh  = (int)(r >> 11);                         // S_=2048 -> shift 11
  int  b   = bh >> 4, h = bh & 15;
  size_t rowoff = ((size_t)(b * S_ + s)) * (3 * DM);
  int    base   = h * DK + 2 * i;
  float q0 = qkv[rowoff + base],          q1 = qkv[rowoff + base + 1];
  float k0 = qkv[rowoff + DM + base],     k1 = qkv[rowoff + DM + base + 1];
  float v0 = qkv[rowoff + 2 * DM + base], v1 = qkv[rowoff + 2 * DM + base + 1];
  float p   = (float)pos[s];
  float inv = __powf(10000.f, -(float)(2 * i) * (1.f / DK));
  float ang = p * inv;
  float c = __cosf(ang), sn = __sinf(ang);
  size_t qoff = ((size_t)bh * S_ + s) * DK + 2 * i;
  Qb[qoff]     = (bf16)(0.125f * (c * q0 - sn * q1));   // fold 1/sqrt(64)
  Qb[qoff + 1] = (bf16)(0.125f * (sn * q0 + c * q1));
  Kb[qoff]     = (bf16)(c * k0 - sn * k1);
  Kb[qoff + 1] = (bf16)(sn * k0 + c * k1);
  size_t voff = ((size_t)bh * DK + 2 * i) * S_ + s;
  Vt[voff]      = (bf16)v0;
  Vt[voff + S_] = (bf16)v1;
}

// ---------------------------------------------------------------------------
// Causal flash attention. One wave (32 thr) per 16-row Q tile per (b,h).
// QK^T and P*V via bf16 WMMA, online softmax with 16-lane shfl reductions.
// Row ownership in C layout: element j on lanes 0-15 -> row j, lanes 16-31 ->
// row j+8, so each lane carries 8 row stats.
// ---------------------------------------------------------------------------
__global__ __launch_bounds__(32) void flash_attn_kernel(
    const bf16* __restrict__ Q, const bf16* __restrict__ Kt,
    const bf16* __restrict__ Vt, bf16* __restrict__ O) {
  int lane = threadIdx.x;
  int qt = blockIdx.x, bh = blockIdx.y;
  const bf16* Qp = Q  + (size_t)bh * S_ * DK;
  const bf16* Kp = Kt + (size_t)bh * S_ * DK;
  const bf16* Vp = Vt + (size_t)bh * DK * S_;   // [dk][s]
  int q0 = qt * 16;

  bf16x16 qa[2];
#pragma unroll
  for (int ks = 0; ks < 2; ++ks) qa[ks] = load_a_frag(Qp, DK, q0, ks * 32, lane);

  f32x8 acc[4] = {};
  float mrow[8], lrow[8];
#pragma unroll
  for (int j = 0; j < 8; ++j) { mrow[j] = NEG_INF; lrow[j] = 0.f; }

  __shared__ __align__(32) bf16 pl[16 * 32];
  int hf = lane >> 4, col = lane & 15;
  int nkt = (q0 + 16 + 31) >> 5;                // causal: k-tiles of 32

  for (int kt = 0; kt < nkt; ++kt) {
    int kc0 = kt * 32;
    f32x8 s0 = {}, s1 = {};
#pragma unroll
    for (int ks = 0; ks < 2; ++ks) {            // Dk = 64 -> two K-steps
      bf16x16 kb0 = load_b_frag(Kp, DK, kc0,      ks * 32, lane);
      bf16x16 kb1 = load_b_frag(Kp, DK, kc0 + 16, ks * 32, lane);
      s0 = wmma_bf16(qa[ks], kb0, s0);
      s1 = wmma_bf16(qa[ks], kb1, s1);
    }
    float p0[8], p1[8], fac[8];
#pragma unroll
    for (int j = 0; j < 8; ++j) {
      int row = q0 + j + hf * 8;
      float v0 = (kc0 + col      <= row) ? s0[j] : NEG_INF;
      float v1 = (kc0 + 16 + col <= row) ? s1[j] : NEG_INF;
      float tm = fmaxf(v0, v1);
#pragma unroll
      for (int msk = 1; msk < 16; msk <<= 1)
        tm = fmaxf(tm, __shfl_xor(tm, msk, 32));
      float mnew = fmaxf(mrow[j], tm);
      fac[j] = __expf(mrow[j] - mnew);
      p0[j]  = __expf(v0 - mnew);
      p1[j]  = __expf(v1 - mnew);
      float ps = p0[j] + p1[j];
#pragma unroll
      for (int msk = 1; msk < 16; msk <<= 1) ps += __shfl_xor(ps, msk, 32);
      lrow[j] = lrow[j] * fac[j] + ps;
      mrow[j] = mnew;
    }
#pragma unroll
    for (int t = 0; t < 4; ++t)
#pragma unroll
      for (int j = 0; j < 8; ++j) acc[t][j] *= fac[j];

    // stage P (16x32) through LDS to convert C-layout -> A-layout
#pragma unroll
    for (int j = 0; j < 8; ++j) {
      int r = j + hf * 8;
      pl[r * 32 + col]      = (bf16)p0[j];
      pl[r * 32 + 16 + col] = (bf16)p1[j];
    }
    __syncthreads();
    bf16x16 pa;
    {
      int r  = lane & 15;
      int kb = hf * 16;
      pa = *(const bf16x16*)&pl[r * 32 + kb];
    }
#pragma unroll
    for (int t = 0; t < 4; ++t) {
      bf16x16 vb = load_b_frag(Vp, S_, t * 16, kc0, lane);
      acc[t] = wmma_bf16(pa, vb, acc[t]);
    }
    __syncthreads();
  }

  // epilogue: O[b, s, h*64+dk] bf16 (input to O-projection GEMM)
  int b = bh >> 4, h = bh & 15;
#pragma unroll
  for (int j = 0; j < 8; ++j) {
    float inv = fast_rcp(lrow[j]);              // v_rcp_f32
    int row = q0 + j + hf * 8;
    size_t base = ((size_t)(b * S_ + row)) * DM + h * DK;
#pragma unroll
    for (int t = 0; t < 4; ++t)
      O[base + t * 16 + col] = (bf16)(acc[t][j] * inv);
  }
}

// ---------------------------------------------------------------------------
// Host orchestration
// ---------------------------------------------------------------------------
extern "C" void kernel_launch(void* const* d_in, const int* in_sizes, int n_in,
                              void* d_out, int out_size, void* d_ws, size_t ws_size,
                              hipStream_t stream) {
  const float* x    = (const float*)d_in[0];
  const int*   pos  = (const int*)  d_in[1];
  const float* qkvw = (const float*)d_in[2];
  const float* ow   = (const float*)d_in[3];
  const float* g1   = (const float*)d_in[4];
  const float* g2   = (const float*)d_in[5];
  const float* w1   = (const float*)d_in[6];
  const float* w2   = (const float*)d_in[7];
  const float* w3   = (const float*)d_in[8];
  float* out = (float*)d_out;

  const size_t NT = (size_t)B_ * S_;            // 8192 tokens
  char* ws = (char*)d_ws;
  size_t off = 0;
  auto take = [&](size_t bytes) {
    char* p = ws + off;
    off += (bytes + 255) & ~(size_t)255;
    return p;
  };
  bf16*  qkvw_b = (bf16*)take((size_t)3 * DM * DM * 2);
  bf16*  ow_b   = (bf16*)take((size_t)DM * DM * 2);
  bf16*  w1_b   = (bf16*)take((size_t)DFF * DM * 2);
  bf16*  w3_b   = (bf16*)take((size_t)DFF * DM * 2);
  bf16*  w2_b   = (bf16*)take((size_t)DM * DFF * 2);
  bf16*  xn_b   = (bf16*)take(NT * DM * 2);     // reused as h1n
  float* qkv_f  = (float*)take(NT * 3 * DM * 4);
  bf16*  hg_b   = (bf16*)qkv_f;                 // reuse (45MB < 96MB)
  bf16*  Qb     = (bf16*)take(NT * DM * 2);
  bf16*  Kb     = (bf16*)take(NT * DM * 2);
  bf16*  Vt     = (bf16*)take(NT * DM * 2);
  bf16*  Ob     = (bf16*)take(NT * DM * 2);
  float* h1     = (float*)take(NT * DM * 4);

  auto cvt = [&](const float* src, bf16* dst, long n) {
    cvt_f32_bf16_kernel<<<(unsigned)((n + 255) / 256), 256, 0, stream>>>(src, dst, n);
  };
  cvt(qkvw, qkvw_b, 3L * DM * DM);
  cvt(ow,   ow_b,   (long)DM * DM);
  cvt(w1,   w1_b,   (long)DFF * DM);
  cvt(w3,   w3_b,   (long)DFF * DM);
  cvt(w2,   w2_b,   (long)DM * DFF);

  // h1 = x + Attn(RMSNorm(x))
  rmsnorm_bf16_kernel<<<(unsigned)NT, 256, 0, stream>>>(x, g1, xn_b);
  wmma_gemm_kernel<false><<<dim3(3 * DM / 128, (unsigned)(NT / 128)), 256, 0, stream>>>(
      xn_b, qkvw_b, qkv_f, nullptr, (int)NT, 3 * DM, DM);
  rope_split_kernel<<<(unsigned)((NT * 512) / 256), 256, 0, stream>>>(
      qkv_f, pos, Qb, Kb, Vt);
  flash_attn_kernel<<<dim3(S_ / 16, B_ * NH), 32, 0, stream>>>(Qb, Kb, Vt, Ob);
  wmma_gemm_kernel<true><<<dim3(DM / 128, (unsigned)(NT / 128)), 256, 0, stream>>>(
      Ob, ow_b, h1, x, (int)NT, DM, DM);

  // out = h1 + W2( silu(W1 h1n) * (W3 h1n) )
  rmsnorm_bf16_kernel<<<(unsigned)NT, 256, 0, stream>>>(h1, g2, xn_b);
  wmma_gemm_swiglu_kernel<<<dim3(DFF / 64, (unsigned)(NT / 128)), 256, 0, stream>>>(
      xn_b, w1_b, w3_b, hg_b, (int)NT, DFF, DM);
  wmma_gemm_kernel<true><<<dim3(DM / 128, (unsigned)(NT / 128)), 256, 0, stream>>>(
      hg_b, w2_b, out, h1, (int)NT, DM, DFF);
}
